// CausalMemoryAttention_3547642987126
// MI455X (gfx1250) — matrix operand order repaired and checked by reading, not compiled, not run on hardware
//
#include <hip/hip_runtime.h>

#define NB 16
#define SS 4096
#define DD 1024
#define CHUNK 256
#define NCHUNK (SS / CHUNK)   // 16
#define DECAY 0.01f
#define INV_SQRT_D 0.03125f   // 1/sqrt(1024)

typedef __attribute__((ext_vector_type(16))) _Float16 v16h;
typedef __attribute__((ext_vector_type(8)))  float    v8f;
typedef __attribute__((ext_vector_type(4)))  float    v4f;

// ---------------------------------------------------------------------------
// WMMA GEMM: Out[16,1024] = A[16,1024] @ (TRANSW ? W^T : W) (+ bias)
// One wave per 16-column output tile; K-loop of 32 per v_wmma_f32_16x16x32_f16.
// A-lane layout (ISA 7.12.2, 16-bit A 16x32): lane L -> row M=L%16,
//   elements e<8 -> K = 8*(L>=16)+e ; e>=8 -> K = 8*(L>=16)+e+8.
// B-lane layout (16-bit B 32x16): lane L -> col N=L%16, same K mapping as A.
// C/D f32 16x16: VGPR r -> (M = r + 8*(L>=16), N = L%16).
// ---------------------------------------------------------------------------
template<bool TRANSW, bool HASBIAS>
__global__ __launch_bounds__(32)
void wmma_gemm16_kernel(const float* __restrict__ A,
                        const float* __restrict__ W,
                        const float* __restrict__ bias,
                        float* __restrict__ Out) {
  const int n0 = blockIdx.x * 16;
  const int L  = threadIdx.x;
  const int hi = L >> 4;
  const int nn = L & 15;          // both B-column and A-row for this lane
  v8f acc = {};
  for (int k0 = 0; k0 < DD; k0 += 32) {
    v16h av, bv;
    const int kbase = hi * 8;
#pragma unroll
    for (int e = 0; e < 16; ++e) {
      const int K = kbase + (e < 8 ? e : e + 8);
      av[e] = (_Float16)A[nn * DD + k0 + K];
    }
    const int koff = hi * 8;
#pragma unroll
    for (int e = 0; e < 16; ++e) {
      const int K = k0 + koff + (e < 8 ? e : e + 8);
      const float w = TRANSW ? W[(n0 + nn) * DD + K] : W[K * DD + n0 + nn];
      bv[e] = (_Float16)w;
    }
    acc = __builtin_amdgcn_wmma_f32_16x16x32_f16(false, av, false, bv,
                                                 (short)0, acc, false, false);
  }
  const float bs = HASBIAS ? bias[n0 + nn] : 0.0f;
#pragma unroll
  for (int r = 0; r < 8; ++r) {
    const int m = r + hi * 8;
    Out[m * DD + n0 + nn] = acc[r] + bs;
  }
}

// qbk[b] = Q[b,:] . bk
__global__ __launch_bounds__(256)
void qdotbk_kernel(const float* __restrict__ Q, const float* __restrict__ bk,
                   float* __restrict__ qbk) {
  const int b = blockIdx.x, t = threadIdx.x;
  float s = 0.0f;
  for (int k = t; k < DD; k += 256) s += Q[b * DD + k] * bk[k];
  __shared__ float red[256];
  red[t] = s; __syncthreads();
  for (int st = 128; st > 0; st >>= 1) {
    if (t < st) red[t] += red[t + st];
    __syncthreads();
  }
  if (t == 0) qbk[b] = red[0];
}

// ---------------------------------------------------------------------------
// Single streaming pass over memory_bank (flash-style online softmax).
// Each block: one (b, S-chunk of 256); 8 waves x 32 rows each.
// Per row: NT-load 1024 floats (v4f x8 per lane), dot with qk, wave-reduce,
// online softmax update of (m, l, acc[1024] in registers). Cross-wave merge
// via LDS ds_add_f32 atomics; partial (m,l,acc) -> workspace.
// ---------------------------------------------------------------------------
__global__ __launch_bounds__(256)
void flash_stream_kernel(const float* __restrict__ Mem,
                         const float* __restrict__ ts,
                         const float* __restrict__ ct,
                         const float* __restrict__ qk,
                         const float* __restrict__ qbk,
                         float* __restrict__ scores,
                         float* __restrict__ pm, float* __restrict__ pl,
                         float* __restrict__ pacc) {
  const int chunk = blockIdx.x;
  const int b = blockIdx.y;
  const int t = threadIdx.x;
  const int w = t >> 5;
  const int l = t & 31;

  __shared__ float smax[8];
  __shared__ float sm_blk;
  __shared__ float sl_blk;
  __shared__ float sacc[DD];
  for (int i = t; i < DD; i += 256) sacc[i] = 0.0f;
  if (t == 0) sl_blk = 0.0f;

  v4f qv[8];
  const v4f* qk4 = (const v4f*)(qk + b * DD);
#pragma unroll
  for (int i = 0; i < 8; ++i) qv[i] = qk4[i * 32 + l];
  const float qb  = qbk[b];
  const float ctb = ct[b];

  float m = -__builtin_inff();
  float lsum = 0.0f;
  v4f acc[8];
#pragma unroll
  for (int i = 0; i < 8; ++i) acc[i] = (v4f)0.0f;

  const int s0 = chunk * CHUNK + w * 32;
  const v4f* Mb = (const v4f*)(Mem + (size_t)b * SS * DD);
  for (int r = 0; r < 32; ++r) {
    const int s = s0 + r;
    const v4f* row = Mb + (size_t)s * (DD / 4);
    if (r + 1 < 32)
      __builtin_prefetch((const void*)(row + (DD / 4) + l * 8), 0, 0);
    v4f rv[8];
#pragma unroll
    for (int i = 0; i < 8; ++i)
      rv[i] = __builtin_nontemporal_load(row + i * 32 + l);
    float d = 0.0f;
#pragma unroll
    for (int i = 0; i < 8; ++i)
      d += rv[i].x * qv[i].x + rv[i].y * qv[i].y +
           rv[i].z * qv[i].z + rv[i].w * qv[i].w;
#pragma unroll
    for (int off = 16; off > 0; off >>= 1) d += __shfl_xor(d, off, 32);
    const float tw = __expf(-DECAY * (ctb - ts[b * SS + s]));
    const float sc = (d + qb) * INV_SQRT_D * tw;
    if (l == 0) scores[b * SS + s] = sc;
    const float mnew  = fmaxf(m, sc);
    const float alpha = __expf(m - mnew);
    const float p     = __expf(sc - mnew);
    lsum = lsum * alpha + p;
#pragma unroll
    for (int i = 0; i < 8; ++i) acc[i] = acc[i] * alpha + rv[i] * p;
    m = mnew;
  }

  if (l == 0) smax[w] = m;
  __syncthreads();
  if (t == 0) {
    float mb = smax[0];
#pragma unroll
    for (int i = 1; i < 8; ++i) mb = fmaxf(mb, smax[i]);
    sm_blk = mb;
  }
  __syncthreads();
  const float alpha = __expf(m - sm_blk);
  if (l == 0) atomicAdd(&sl_blk, lsum * alpha);
#pragma unroll
  for (int i = 0; i < 8; ++i) {
    const int dc = i * 128 + l * 4;
    atomicAdd(&sacc[dc + 0], acc[i].x * alpha);
    atomicAdd(&sacc[dc + 1], acc[i].y * alpha);
    atomicAdd(&sacc[dc + 2], acc[i].z * alpha);
    atomicAdd(&sacc[dc + 3], acc[i].w * alpha);
  }
  __syncthreads();
  const int pidx = b * NCHUNK + chunk;
  if (t == 0) { pm[pidx] = sm_blk; pl[pidx] = sl_blk; }
  float* pa = pacc + (size_t)pidx * DD;
  for (int i = t; i < DD; i += 256) pa[i] = sacc[i];
}

// Merge the 16 chunk-partials per batch: global m, l, and a[b,:] = attn @ M
__global__ __launch_bounds__(256)
void merge_kernel(const float* __restrict__ pm, const float* __restrict__ pl,
                  const float* __restrict__ pacc,
                  float* __restrict__ avec,
                  float* __restrict__ mbo, float* __restrict__ lbo) {
  const int b = blockIdx.x, t = threadIdx.x;
  __shared__ float sf[NCHUNK];
  __shared__ float s_inv;
  if (t == 0) {
    float mb = pm[b * NCHUNK];
    for (int c = 1; c < NCHUNK; ++c) mb = fmaxf(mb, pm[b * NCHUNK + c]);
    float lt = 0.0f;
    for (int c = 0; c < NCHUNK; ++c) {
      const float f = __expf(pm[b * NCHUNK + c] - mb);
      sf[c] = f;
      lt += f * pl[b * NCHUNK + c];
    }
    mbo[b] = mb; lbo[b] = lt; s_inv = 1.0f / lt;
  }
  __syncthreads();
  const float inv = s_inv;
  for (int d = t; d < DD; d += 256) {
    float a = 0.0f;
    for (int c = 0; c < NCHUNK; ++c)
      a += sf[c] * pacc[(size_t)(b * NCHUNK + c) * DD + d];
    avec[b * DD + d] = a * inv;
  }
}

// Normalized attention weights output: exp(score - m)/l
__global__ __launch_bounds__(256)
void attn_out_kernel(const float* __restrict__ scores,
                     const float* __restrict__ mb, const float* __restrict__ lb,
                     float* __restrict__ outw) {
  const int b = blockIdx.y;
  const int s = blockIdx.x * 256 + threadIdx.x;
  outw[b * SS + s] = __expf(scores[b * SS + s] - mb[b]) / lb[b];
}

extern "C" void kernel_launch(void* const* d_in, const int* in_sizes, int n_in,
                              void* d_out, int out_size, void* d_ws, size_t ws_size,
                              hipStream_t stream) {
  (void)in_sizes; (void)n_in; (void)out_size; (void)ws_size;
  const float* query = (const float*)d_in[0];
  const float* mem   = (const float*)d_in[1];
  const float* ts    = (const float*)d_in[2];
  const float* ct    = (const float*)d_in[3];
  const float* Wq    = (const float*)d_in[4];
  const float* bq    = (const float*)d_in[5];
  const float* Wk    = (const float*)d_in[6];
  const float* bk    = (const float*)d_in[7];
  const float* Wv    = (const float*)d_in[8];
  const float* bv    = (const float*)d_in[9];

  float* out      = (float*)d_out;
  float* attended = out;            // [16,1024]
  float* attnw    = out + NB * DD;  // [16,4096]

  float* ws    = (float*)d_ws;
  float* Qws   = ws;  ws += NB * DD;
  float* qkws  = ws;  ws += NB * DD;
  float* qbk   = ws;  ws += NB;
  float* scor  = ws;  ws += NB * SS;
  float* pm    = ws;  ws += NB * NCHUNK;
  float* pl    = ws;  ws += NB * NCHUNK;
  float* pacc  = ws;  ws += (size_t)NB * NCHUNK * DD;
  float* mb    = ws;  ws += NB;
  float* lb    = ws;  ws += NB;
  float* avec  = ws;  ws += NB * DD;

  // Q = query @ Wq^T + bq      (WMMA)
  wmma_gemm16_kernel<true, true><<<DD / 16, 32, 0, stream>>>(query, Wq, bq, Qws);
  // qk = Q @ Wk                (WMMA)  [folds K-projection into a vector]
  wmma_gemm16_kernel<false, false><<<DD / 16, 32, 0, stream>>>(Qws, Wk, nullptr, qkws);
  // qbk[b] = Q[b] . bk
  qdotbk_kernel<<<NB, 256, 0, stream>>>(Qws, bk, qbk);
  // single pass over memory_bank: scores + online-softmax-weighted attn@M
  flash_stream_kernel<<<dim3(NCHUNK, NB), 256, 0, stream>>>(mem, ts, ct, qkws,
                                                            qbk, scor, pm, pl, pacc);
  // merge split-S partials
  merge_kernel<<<NB, 256, 0, stream>>>(pm, pl, pacc, avec, mb, lb);
  // attn_weights output
  attn_out_kernel<<<dim3(SS / 256, NB), 256, 0, stream>>>(scor, mb, lb, attnw);
  // attended = (attn@M) @ Wv^T + bv   (WMMA)  [folds V-projection]
  wmma_gemm16_kernel<true, true><<<DD / 16, 32, 0, stream>>>(avec, Wv, bv, attended);
}
